// NativeAttention_26594437497704
// MI455X (gfx1250) — matrix operand order repaired and checked by reading, not compiled, not run on hardware
//
#include <hip/hip_runtime.h>
#include <hip/hip_bf16.h>

#define D_MODEL   2048
#define SEQ       2048
#define BATCH     4
#define NUM_HEADS 16
#define HEAD_DIM  128
#define MROWS     (BATCH*SEQ)   // 8192

typedef __bf16 bf16_t;
typedef __attribute__((ext_vector_type(16))) __bf16 v16bf;
typedef __attribute__((ext_vector_type(8)))  __bf16 v8bf;
typedef __attribute__((ext_vector_type(8)))  float  v8f;
typedef __attribute__((ext_vector_type(4)))  int    v4i;

#define AS1 __attribute__((address_space(1)))
#define AS3 __attribute__((address_space(3)))

#if __has_builtin(__builtin_amdgcn_global_load_async_to_lds_b128) && \
    __has_builtin(__builtin_amdgcn_s_wait_asynccnt)
#define HAVE_ASYNC_LDS 1
#else
#define HAVE_ASYNC_LDS 0
#endif

static __device__ __forceinline__ v16bf cat8(v8bf lo, v8bf hi) {
  return __builtin_shufflevector(lo, hi, 0,1,2,3,4,5,6,7,8,9,10,11,12,13,14,15);
}
static __device__ __forceinline__ v8f zero8() {
  v8f z = {0.f,0.f,0.f,0.f,0.f,0.f,0.f,0.f};
  return z;
}
static __device__ __forceinline__ v8f wmma_bf16(v16bf a, v16bf b, v8f c) {
  return __builtin_amdgcn_wmma_f32_16x16x32_bf16(false, a, false, b, (short)0, c, false, false);
}

// 16-byte global -> LDS copy: async (ASYNCcnt-tracked) when available.
static __device__ __forceinline__ void async_copy_b128(const bf16_t* g, bf16_t* l) {
#if HAVE_ASYNC_LDS
  __builtin_amdgcn_global_load_async_to_lds_b128(
      (AS1 v4i*)g, (AS3 v4i*)l, 0, 0);
#else
  *(v8bf*)l = *(const v8bf*)g;
#endif
}
static __device__ __forceinline__ void wait_async() {
#if HAVE_ASYNC_LDS
  __builtin_amdgcn_s_wait_asynccnt(0);
#endif
}

// ---------------------------------------------------------------------------
// f32 -> bf16 conversion (grid-stride)
// ---------------------------------------------------------------------------
__global__ void cvt_f32_bf16(const float* __restrict__ src, bf16_t* __restrict__ dst, int n) {
  int i = blockIdx.x * blockDim.x + threadIdx.x;
  int stride = gridDim.x * blockDim.x;
  for (; i < n; i += stride) dst[i] = (bf16_t)src[i];
}

// ---------------------------------------------------------------------------
// C[M=8192, N=2048] = X[M,K=2048] @ W[N,K]^T   (both row-major, bf16)
// WG = 256 threads (8 waves), tile 128x128, K-step 32.
// Tiles double-buffered in LDS via async b128 fills; fragments read from LDS.
// LDS row stride 40 elems (20 dword banks) -> conflict-free fragment reads.
// ---------------------------------------------------------------------------
#define PADX 40

__global__ __launch_bounds__(256) void gemm_bf16_wmma(
    const bf16_t* __restrict__ X, const bf16_t* __restrict__ W,
    bf16_t* __restrict__ outH, bf16_t* __restrict__ outT,
    float* __restrict__ outF, int rowMajorF)
{
  __shared__ bf16_t ldsX[2][128 * PADX];
  __shared__ bf16_t ldsW[2][128 * PADX];

  const int tid    = threadIdx.x;
  const int lane   = tid & 31;
  const int w      = tid >> 5;
  const int waveM  = w & 3;
  const int waveN  = w >> 2;
  const int tileN  = blockIdx.x * 128;
  const int tileM  = blockIdx.y * 128;
  const int l15    = lane & 15;
  const int hiHalf = lane >> 4;   // 0 or 1

  auto stage = [&](int buf, int kk) {
    #pragma unroll
    for (int c = tid; c < 512; c += 256) {       // 128 rows x 4 chunks of 8 elems
      const int row = c >> 2;
      const int c8  = (c & 3) * 8;
      async_copy_b128(X + (size_t)(tileM + row) * D_MODEL + kk + c8,
                      &ldsX[buf][row * PADX + c8]);
      async_copy_b128(W + (size_t)(tileN + row) * D_MODEL + kk + c8,
                      &ldsW[buf][row * PADX + c8]);
    }
  };

  v8f acc[2][4];
  #pragma unroll
  for (int i = 0; i < 2; ++i)
    #pragma unroll
    for (int j = 0; j < 4; ++j) acc[i][j] = zero8();

  stage(0, 0);
  const int nk = D_MODEL / 32;   // 64

  #pragma unroll 1
  for (int ik = 0; ik < nk; ++ik) {
    wait_async();
    __syncthreads();                       // fills visible; prev reads done
    if (ik + 1 < nk) stage((ik + 1) & 1, (ik + 1) * 32);  // overlap next fill

    const bf16_t* xs = ldsX[ik & 1];
    const bf16_t* wt = ldsW[ik & 1];

    // A fragments: 16x32, lane = row, 16 consecutive K at kbase=hiHalf*16
    v16bf a[2];
    #pragma unroll
    for (int mi = 0; mi < 2; ++mi) {
      const bf16_t* p = xs + (waveM*32 + mi*16 + l15) * PADX + hiHalf*16;
      a[mi] = cat8(*(const v8bf*)p, *(const v8bf*)(p + 8));
    }
    // B fragments: 32x16, lane = column n, 16 consecutive K from W row n
    #pragma unroll
    for (int ni = 0; ni < 4; ++ni) {
      const bf16_t* p = wt + (waveN*64 + ni*16 + l15) * PADX + hiHalf*16;
      v16bf b = cat8(*(const v8bf*)p, *(const v8bf*)(p + 8));
      acc[0][ni] = wmma_bf16(a[0], b, acc[0][ni]);
      acc[1][ni] = wmma_bf16(a[1], b, acc[1][ni]);
    }
  }

  // Epilogue: C layout: VGPR r -> row (hiHalf ? 8+r : r), col = l15
  #pragma unroll
  for (int mi = 0; mi < 2; ++mi) {
    #pragma unroll
    for (int ni = 0; ni < 4; ++ni) {
      #pragma unroll
      for (int r = 0; r < 8; ++r) {
        const int m = tileM + waveM*32 + mi*16 + (hiHalf ? 8 + r : r);
        const int n = tileN + waveN*64 + ni*16 + l15;
        const float v = acc[mi][ni][r];
        const int b = m >> 11, s = m & (SEQ - 1);
        const int h = n >> 7,  d = n & (HEAD_DIM - 1);
        const size_t headIdx = (((size_t)b*NUM_HEADS + h)*SEQ + s)*HEAD_DIM + d;
        if (outH) outH[headIdx] = (bf16_t)v;
        if (outT) outT[(((size_t)b*NUM_HEADS + h)*HEAD_DIM + d)*SEQ + s] = (bf16_t)v;
        if (outF) outF[rowMajorF ? ((size_t)m * D_MODEL + n) : headIdx] = v;
      }
    }
  }
}

// ---------------------------------------------------------------------------
// Causal flash attention.  grid = (qTiles=16, b*h=64), block = 256 (8 waves).
// K block (32x128) and V^T block (128x32) staged in LDS (double-buffered,
// async fill overlapped with compute).  P reshaped C->A layout through LDS.
// ---------------------------------------------------------------------------
#define PADK 136   // 128 + 8 (68 dword stride)
#define PADV 40    // 32 + 8  (20 dword stride)

__global__ __launch_bounds__(256) void attn_wmma(
    const bf16_t* __restrict__ Qbf, const bf16_t* __restrict__ Kbf,
    const bf16_t* __restrict__ Vt,  bf16_t* __restrict__ Yh)
{
  __shared__ bf16_t ldsK[2][32 * PADK];    // 17408 B
  __shared__ bf16_t ldsV[2][128 * PADV];   // 20480 B
  __shared__ bf16_t ldsP[8 * 16 * PADV];   // 10240 B

  const int tid    = threadIdx.x;
  const int lane   = tid & 31;
  const int w      = tid >> 5;
  const int l15    = lane & 15;
  const int hiHalf = lane >> 4;
  const int bh     = blockIdx.y;
  const int b      = bh >> 4, h = bh & 15;
  const int tileQ  = blockIdx.x;
  const int qRowBase = tileQ*128 + w*16;

  const bf16_t* Qp = Qbf + (size_t)bh * SEQ * HEAD_DIM;
  const bf16_t* Kp = Kbf + (size_t)bh * SEQ * HEAD_DIM;
  const bf16_t* Vp = Vt  + (size_t)bh * HEAD_DIM * SEQ;

  auto stageKV = [&](int buf, int k0) {
    #pragma unroll
    for (int c = tid; c < 512; c += 256) {       // K: 32 rows x 16 chunks
      const int row = c >> 4;
      const int c8  = (c & 15) * 8;
      async_copy_b128(Kp + (size_t)(k0 + row) * HEAD_DIM + c8,
                      &ldsK[buf][row * PADK + c8]);
    }
    #pragma unroll
    for (int c = tid; c < 512; c += 256) {       // V^T: 128 rows x 4 chunks
      const int row = c >> 2;
      const int c8  = (c & 3) * 8;
      async_copy_b128(Vp + (size_t)row * SEQ + k0 + c8,
                      &ldsV[buf][row * PADV + c8]);
    }
  };

  // Preload Q fragments (16 rows x 128 d as 4 A-frags of 16x32)
  v16bf aQ[4];
  {
    const int qr = qRowBase + l15;
    #pragma unroll
    for (int d0 = 0; d0 < 4; ++d0) {
      const bf16_t* p = Qp + (size_t)qr * HEAD_DIM + d0*32 + hiHalf*16;
      aQ[d0] = cat8(*(const v8bf*)p, *(const v8bf*)(p + 8));
    }
  }

  v8f o[8];
  #pragma unroll
  for (int i = 0; i < 8; ++i) o[i] = zero8();
  float rm[8], rl[8];
  #pragma unroll
  for (int r = 0; r < 8; ++r) { rm[r] = -3.0e38f; rl[r] = 0.0f; }

  const float scale = 0.08838834764831845f; // 1/sqrt(128)
  const int nkb = tileQ*4 + 4;              // 32-key blocks in causal span
  bf16_t* pl = ldsP + w * (16 * PADV);

  stageKV(0, 0);

  #pragma unroll 1
  for (int kb = 0; kb < nkb; ++kb) {
    wait_async();
    __syncthreads();
    if (kb + 1 < nkb) stageKV((kb + 1) & 1, (kb + 1) * 32);

    const bf16_t* ks = ldsK[kb & 1];
    const bf16_t* vs = ldsV[kb & 1];
    const int k0 = kb * 32;

    // ---- scores: S = Q . K^T, two 16x16 tiles over this 32-key block
    v8f s0 = zero8(), s1 = zero8();
    #pragma unroll
    for (int d0 = 0; d0 < 4; ++d0) {
      const int db = d0*32 + hiHalf*16;
      const bf16_t* p0 = ks + (l15) * PADK + db;
      v16bf bk0 = cat8(*(const v8bf*)p0, *(const v8bf*)(p0 + 8));
      s0 = wmma_bf16(aQ[d0], bk0, s0);
      const bf16_t* p1 = ks + (16 + l15) * PADK + db;
      v16bf bk1 = cat8(*(const v8bf*)p1, *(const v8bf*)(p1 + 8));
      s1 = wmma_bf16(aQ[d0], bk1, s1);
    }

    // ---- causal mask + online softmax (half-wave reductions via shfl_xor)
    const int c0 = k0 + l15, c1 = c0 + 16;
    float fcorr[8];
    #pragma unroll
    for (int r = 0; r < 8; ++r) {
      const int row = qRowBase + (hiHalf ? 8 + r : r);
      float x0 = s0[r] * scale; if (c0 > row) x0 = -3.0e38f;
      float x1 = s1[r] * scale; if (c1 > row) x1 = -3.0e38f;
      float mloc = fmaxf(x0, x1);
      #pragma unroll
      for (int msk = 8; msk >= 1; msk >>= 1)
        mloc = fmaxf(mloc, __shfl_xor(mloc, msk, 32));
      const float mnew = fmaxf(rm[r], mloc);
      const float f  = __expf(rm[r] - mnew);
      const float p0 = __expf(x0 - mnew);
      const float p1 = __expf(x1 - mnew);
      s0[r] = p0; s1[r] = p1;
      float ls = p0 + p1;
      #pragma unroll
      for (int msk = 8; msk >= 1; msk >>= 1)
        ls += __shfl_xor(ls, msk, 32);
      rl[r] = rl[r] * f + ls;
      rm[r] = mnew;
      fcorr[r] = f;
    }
    #pragma unroll
    for (int ni = 0; ni < 8; ++ni)
      #pragma unroll
      for (int r = 0; r < 8; ++r) o[ni][r] *= fcorr[r];

    // ---- reshape P: C-layout -> A-layout via LDS
    #pragma unroll
    for (int r = 0; r < 8; ++r) {
      const int row = hiHalf ? 8 + r : r;
      pl[row*PADV + l15]      = (bf16_t)s0[r];
      pl[row*PADV + l15 + 16] = (bf16_t)s1[r];
    }
    __syncthreads();
    const bf16_t* ap = pl + l15*PADV + hiHalf*16;
    v16bf aP = cat8(*(const v8bf*)ap, *(const v8bf*)(ap + 8));

    // ---- O += P . V   (V^T tile in LDS gives contiguous B-fragments)
    #pragma unroll
    for (int ni = 0; ni < 8; ++ni) {
      const bf16_t* vp = vs + (ni*16 + l15) * PADV + hiHalf*16;
      v16bf bv = cat8(*(const v8bf*)vp, *(const v8bf*)(vp + 8));
      o[ni] = wmma_bf16(aP, bv, o[ni]);
    }
  }

  // ---- normalize and store heads-merged layout [b, s, h*128+d] (bf16)
  float inv[8];
  #pragma unroll
  for (int r = 0; r < 8; ++r) inv[r] = 1.0f / rl[r];
  #pragma unroll
  for (int ni = 0; ni < 8; ++ni) {
    #pragma unroll
    for (int r = 0; r < 8; ++r) {
      const int row = qRowBase + (hiHalf ? 8 + r : r);
      const int d = ni*16 + l15;
      Yh[((size_t)b*SEQ + row)*D_MODEL + h*HEAD_DIM + d] = (bf16_t)(o[ni][r] * inv[r]);
    }
  }
}

// ---------------------------------------------------------------------------
extern "C" void kernel_launch(void* const* d_in, const int* in_sizes, int n_in,
                              void* d_out, int out_size, void* d_ws, size_t ws_size,
                              hipStream_t stream) {
  const float* x  = (const float*)d_in[0];
  const float* Wq = (const float*)d_in[1];
  const float* Wk = (const float*)d_in[2];
  const float* Wv = (const float*)d_in[3];
  const float* Wo = (const float*)d_in[4];

  const size_t NX = (size_t)MROWS * D_MODEL;    // 16,777,216
  const size_t NW = (size_t)D_MODEL * D_MODEL;  //  4,194,304

  bf16_t* ws  = (bf16_t*)d_ws;
  bf16_t* Xbf = ws;
  bf16_t* Wqb = Xbf + NX;
  bf16_t* Wkb = Wqb + NW;
  bf16_t* Wvb = Wkb + NW;
  bf16_t* Wob = Wvb + NW;
  bf16_t* Qbf = Wob + NW;
  bf16_t* Kb  = Qbf + NX;
  bf16_t* Vtb = Kb  + NX;
  bf16_t* Yhb = Vtb + NX;

  float* y  = (float*)d_out;   // [b, s, d_model]
  float* kO = y  + NX;         // [b, h, s, d]
  float* vO = kO + NX;         // [b, h, s, d]

  cvt_f32_bf16<<<4096, 256, 0, stream>>>(x,  Xbf, (int)NX);
  cvt_f32_bf16<<<2048, 256, 0, stream>>>(Wq, Wqb, (int)NW);
  cvt_f32_bf16<<<2048, 256, 0, stream>>>(Wk, Wkb, (int)NW);
  cvt_f32_bf16<<<2048, 256, 0, stream>>>(Wv, Wvb, (int)NW);
  cvt_f32_bf16<<<2048, 256, 0, stream>>>(Wo, Wob, (int)NW);

  dim3 g(D_MODEL/128, MROWS/128), blk(256);
  // Q: bf16 head layout only
  gemm_bf16_wmma<<<g, blk, 0, stream>>>(Xbf, Wqb, Qbf, nullptr, nullptr, 0);
  // K: bf16 head layout + f32 output
  gemm_bf16_wmma<<<g, blk, 0, stream>>>(Xbf, Wkb, Kb, nullptr, kO, 0);
  // V: bf16 transposed layout + f32 output
  gemm_bf16_wmma<<<g, blk, 0, stream>>>(Xbf, Wvb, nullptr, Vtb, vO, 0);

  attn_wmma<<<dim3(SEQ/128, BATCH*NUM_HEADS), 256, 0, stream>>>(Qbf, Kb, Vtb, Yhb);

  // y = Yh @ Wo^T, f32 row-major
  gemm_bf16_wmma<<<g, blk, 0, stream>>>(Yhb, Wob, nullptr, nullptr, y, 1);
}
